// Encoder_59485297050242
// MI455X (gfx1250) — matrix operand order repaired and checked by reading, not compiled, not run on hardware
//
#include <hip/hip_runtime.h>
#include <hip/hip_bf16.h>
#include <cstdint>

typedef __attribute__((ext_vector_type(16))) _Float16 v16h;
typedef __attribute__((ext_vector_type(8)))  _Float16 v8h;
typedef __attribute__((ext_vector_type(8)))  float    v8f;
typedef __attribute__((ext_vector_type(4)))  unsigned int v4u;
typedef __attribute__((ext_vector_type(8)))  int v8i;
typedef __attribute__((ext_vector_type(4)))  int v4i;

union H16U { v16h v; v8h h[2]; };

__device__ __forceinline__ float sigmf(float x) { return 1.0f / (1.0f + __expf(-x)); }

// ---------------- mask kernels ----------------
__global__ void mask_from_input(const float* __restrict__ x, int* __restrict__ m,
                                int rows, int F) {
    int i = blockIdx.x * blockDim.x + threadIdx.x;
    if (i >= rows) return;
    const float* p = x + (size_t)i * F;
    int any = 0;
    for (int j = 0; j < F; ++j) any |= (p[j] != 0.0f);
    m[i] = any;
}

__global__ void mask_down(const int* __restrict__ src, int* __restrict__ dst, int Tn) {
    int i = blockIdx.x * blockDim.x + threadIdx.x;
    if (i >= 32 * Tn) return;
    int b = i / Tn, t = i - b * Tn;
    dst[i] = src[b * 2 * Tn + 2 * t] | src[b * 2 * Tn + 2 * t + 1];
}

// ---------------- convert / pack kernels ----------------
__global__ void pad_x_f16(const float* __restrict__ x, _Float16* __restrict__ o,
                          int rows, int F, int Kp) {
    size_t i = (size_t)blockIdx.x * blockDim.x + threadIdx.x;
    if (i >= (size_t)rows * Kp) return;
    int r = (int)(i / Kp), k = (int)(i - (size_t)r * Kp);
    o[i] = (k < F) ? (_Float16)x[(size_t)r * F + k] : (_Float16)0.0f;
}

// Pack a [K x 1536] f32 weight matrix into the exact per-lane WMMA B-matrix
// register image: img[nt][chunk][lane][e], 16 halves per lane per chunk.
// B 32x16 layout (wave32): lanes 0-15 hold K=0..7,16..23, lanes 16-31 hold K=8..15,24..31,
// column N = lane%16.
__global__ void build_bimg(const float* __restrict__ w, _Float16* __restrict__ img,
                           int kchunks, int K) {
    size_t i = (size_t)blockIdx.x * blockDim.x + threadIdx.x;
    size_t tot = (size_t)96 * kchunks * 32 * 16;
    if (i >= tot) return;
    int e    = (int)(i & 15);
    int lane = (int)((i >> 4) & 31);
    int c    = (int)((i >> 9) % (size_t)kchunks);
    int nt   = (int)(i / ((size_t)kchunks << 9));
    int n = nt * 16 + (lane & 15);
    int k = c * 32 + ((e & 8) ? 16 : 0) + ((lane & 16) ? 8 : 0) + (e & 7);
    img[i] = (k < K) ? (_Float16)w[(size_t)k * 1536 + n] : (_Float16)0.0f;
}

// ---------------- xproj GEMM: xz = x @ k + b0 ----------------
// A: row-major f16 [M x lda]; B: packed image; out: f16 [M x 1536].
// One wave computes TWO adjacent 16x16 C tiles (A fragment reused), 8 waves/block.
__global__ __launch_bounds__(256) void wmma_gemm_xz(
    const _Float16* __restrict__ A, int lda, int kchunks,
    const _Float16* __restrict__ bimg, const float* __restrict__ bias,
    _Float16* __restrict__ out, int M) {
    const int lane = threadIdx.x & 31;
    const int wave = threadIdx.x >> 5;
    const int tile = blockIdx.x * 8 + wave;       // each tile = 2 ntiles
    const int mt = tile / 48;
    const int nt = (tile - mt * 48) * 2;
    if (mt * 16 >= M) return;
    const int row = mt * 16 + (lane & 15);
    const int ksel = (lane & 16) ? 8 : 0;         // A 16x32 layout K-octet select
    v8f acc0 = {}, acc1 = {};
    for (int c = 0; c < kchunks; ++c) {
        H16U a, b0, b1;
        const v8h* pa = (const v8h*)(A + (size_t)row * lda + c * 32 + ksel);
        a.h[0] = pa[0];           // K octets 0,2 (or 1,3)
        a.h[1] = pa[2];
        const v8h* pb0 = (const v8h*)(bimg + (((size_t)nt * kchunks + c) * 32 + lane) * 16);
        const v8h* pb1 = (const v8h*)(bimg + (((size_t)(nt + 1) * kchunks + c) * 32 + lane) * 16);
        b0.h[0] = pb0[0]; b0.h[1] = pb0[1];
        b1.h[0] = pb1[0]; b1.h[1] = pb1[1];
        acc0 = __builtin_amdgcn_wmma_f32_16x16x32_f16(false, a.v, false, b0.v,
                                                      (short)0, acc0, false, false);
        acc1 = __builtin_amdgcn_wmma_f32_16x16x32_f16(false, a.v, false, b1.v,
                                                      (short)0, acc1, false, false);
    }
    const int mbase = mt * 16 + ((lane & 16) ? 8 : 0);
    const int n0 = nt * 16 + (lane & 15);
    const float bn0 = bias[n0];
    const float bn1 = bias[n0 + 16];
#pragma unroll
    for (int i = 0; i < 8; ++i) {
        out[(size_t)(mbase + i) * 1536 + n0]      = (_Float16)(acc0[i] + bn0);
        out[(size_t)(mbase + i) * 1536 + n0 + 16] = (_Float16)(acc1[i] + bn1);
    }
}

// ---------------- grid-wide barrier (persistent kernel) ----------------
__device__ __forceinline__ void gsync(int* counter, int& target, int nblk) {
    __syncthreads();
    target += nblk;
    if (threadIdx.x == 0) {
        __threadfence();
        atomicAdd(counter, 1);
        while (__hip_atomic_load(counter, __ATOMIC_RELAXED, __HIP_MEMORY_SCOPE_AGENT) < target)
            __builtin_amdgcn_s_sleep(1);
    }
    __syncthreads();
    __threadfence();
}

// ---------------- persistent BiGRU scan ----------------
// 64 blocks x 64 threads. blockIdx: dir = b>>5, hidden-col block nb = b&31.
// Wave w handles batch rows [16w,16w+16). Recurrent weights for the 3 gate
// slices {nb, 32+nb, 64+nb} live in LDS in WMMA B-image layout (48KB),
// filled by the Tensor Data Mover (2D tile: 3 rows of 16KB, 512KB stride).
__global__ __launch_bounds__(64) void bigru_scan(
    const _Float16* __restrict__ rimg,   // [2][96][16][32][16]
    const float* __restrict__ b1f, const float* __restrict__ b1b,
    const _Float16* __restrict__ xz,     // [2][32*T][1536]
    const int* __restrict__ mask,        // [32*T]
    _Float16* __restrict__ ys16,         // [32*T][1024]
    float* __restrict__ out32,           // nullptr or d_out for last layer
    _Float16* __restrict__ h16,          // [2 phase][2 dir][32][512]
    float* __restrict__ h32,             // [2 dir][32][512]
    int T, int* counter) {
    __shared__ _Float16 lds_b[3][16][32][16];   // 48 KB
    const int nb  = blockIdx.x & 31;
    const int dir = blockIdx.x >> 5;
    const int tid = threadIdx.x;
    const int lane = tid & 31;
    const int wave = tid >> 5;
    const int nblk = (int)gridDim.x;
    const float* bias1 = dir ? b1b : b1f;
    const _Float16* xzd = xz + (size_t)dir * 32 * T * 1536;

#if __has_builtin(__builtin_amdgcn_tensor_load_to_lds)
    if (wave == 0) {
        // D# describing: 3 rows (gate slices) x 2048 8-byte elements, row
        // stride 65536 elements (= 32 tiles * 16KB), contiguous into LDS.
        const _Float16* gsrc = rimg + ((size_t)(dir * 96) + nb) * 8192;  // gate-0 slice
        unsigned long long ga = (unsigned long long)(size_t)gsrc;
        unsigned int lds_lo = (unsigned int)(size_t)(&lds_b[0][0][0][0]);
        v4u g0; v8i g1; v4i g2 = {}; v4i g3 = {}; v8i g4 = {};
        g0[0] = 1u;                                     // count=1, user desc
        g0[1] = lds_lo;                                 // lds_addr
        g0[2] = (unsigned int)ga;                       // global_addr[31:0]
        g0[3] = (unsigned int)(ga >> 32) | (2u << 30);  // global_addr[56:32] | type=2
        g1[0] = (int)(3u << 16);        // data_size = 8B; wg_mask = 0
        g1[1] = (int)(2048u << 16);     // tensor_dim0[15:0] = 2048
        g1[2] = (int)(3u << 16);        // tensor_dim0 hi = 0 | tensor_dim1[15:0] = 3
        g1[3] = (int)(2048u << 16);     // tensor_dim1 hi = 0 | tile_dim0 = 2048
        g1[4] = 3;                      // tile_dim1 = 3, tile_dim2 = 0
        g1[5] = 65536;                  // tensor_dim0_stride[31:0]
        g1[6] = 0;                      // stride hi | tensor_dim1_stride lo
        g1[7] = 0;
        __builtin_amdgcn_tensor_load_to_lds(g0, g1, g2, g3, g4, 0);
        __builtin_amdgcn_s_wait_tensorcnt(0);
    }
#else
    for (int g = 0; g < 3; ++g) {
        const uint4* s = (const uint4*)(rimg + ((size_t)(dir * 96 + g * 32 + nb)) * 8192);
        uint4* d = (uint4*)(&lds_b[g][0][0][0]);
        for (int i = tid; i < 1024; i += 64) d[i] = s[i];
    }
#endif
    // Zero our 16 hidden columns (both ping-pong phases + f32 master).
    for (int k = tid; k < 512; k += 64) {
        int b = k >> 4, c = k & 15;
        int col = nb * 16 + c;
        h32[dir * 16384 + b * 512 + col] = 0.0f;
        h16[((size_t)(0 * 2 + dir) * 32 + b) * 512 + col] = (_Float16)0.0f;
        h16[((size_t)(1 * 2 + dir) * 32 + b) * 512 + col] = (_Float16)0.0f;
    }
    __syncthreads();
    int target = 0;
    gsync(counter, target, nblk);   // all h zeros visible everywhere

    const int row  = wave * 16 + (lane & 15);
    const int ksel = (lane & 16) ? 8 : 0;
    const int col  = nb * 16 + (lane & 15);
    const float bz = bias1[col];
    const float br = bias1[512 + col];
    const float bh = bias1[1024 + col];
    const int mb_base = wave * 16 + ((lane & 16) ? 8 : 0);

    for (int s = 0; s < T; ++s) {
        const int t = dir ? (T - 1 - s) : s;
        const int phase = s & 1;
        const _Float16* hsrc = h16 + ((size_t)(phase * 2 + dir) * 32) * 512;
        v8f az = {}, ar = {}, ah = {};
#pragma unroll 2
        for (int c = 0; c < 16; ++c) {
            H16U a, b0u, b1u, b2u;
            const v8h* pa = (const v8h*)(hsrc + (size_t)row * 512 + c * 32 + ksel);
            a.h[0] = pa[0]; a.h[1] = pa[2];
            const v8h* p0 = (const v8h*)&lds_b[0][c][lane][0];
            const v8h* p1 = (const v8h*)&lds_b[1][c][lane][0];
            const v8h* p2 = (const v8h*)&lds_b[2][c][lane][0];
            b0u.h[0] = p0[0]; b0u.h[1] = p0[1];
            b1u.h[0] = p1[0]; b1u.h[1] = p1[1];
            b2u.h[0] = p2[0]; b2u.h[1] = p2[1];
            az = __builtin_amdgcn_wmma_f32_16x16x32_f16(false, a.v, false, b0u.v, (short)0, az, false, false);
            ar = __builtin_amdgcn_wmma_f32_16x16x32_f16(false, a.v, false, b1u.v, (short)0, ar, false, false);
            ah = __builtin_amdgcn_wmma_f32_16x16x32_f16(false, a.v, false, b2u.v, (short)0, ah, false, false);
        }
        _Float16* hdst = h16 + ((size_t)((1 - phase) * 2 + dir) * 32) * 512;
#pragma unroll
        for (int i = 0; i < 8; ++i) {
            const int mb = mb_base + i;
            const size_t xb = ((size_t)mb * T + t) * 1536;
            float xzz = (float)xzd[xb + col];
            float xzr = (float)xzd[xb + 512 + col];
            float xzh = (float)xzd[xb + 1024 + col];
            float z  = sigmf(xzz + az[i] + bz);
            float r  = sigmf(xzr + ar[i] + br);
            float hh = tanhf(xzh + r * (ah[i] + bh));
            float hold = h32[dir * 16384 + mb * 512 + col];
            int mv = mask[mb * T + t];
            float hn = mv ? (z * hold + (1.0f - z) * hh) : hold;
            h32[dir * 16384 + mb * 512 + col] = hn;
            hdst[(size_t)mb * 512 + col] = (_Float16)hn;
            const size_t yi = ((size_t)mb * T + t) * 1024 + dir * 512 + col;
            ys16[yi] = mv ? (_Float16)hn : (_Float16)0.0f;
            if (out32) out32[yi] = mv ? hn : 0.0f;
        }
        if (s + 1 < T) {
            const int t2 = dir ? (T - 2 - s) : (s + 1);
            __builtin_prefetch(&xzd[((size_t)mb_base * T + t2) * 1536 + col], 0, 1);
        }
        gsync(counter, target, nblk);
    }
}

// ---------------- final hidden-state assembly ----------------
__global__ void assemble_hidden(const float* __restrict__ h32, float* __restrict__ out,
                                size_t off) {
    int i = blockIdx.x * blockDim.x + threadIdx.x;
    if (i >= 32 * 1024) return;
    int b = i >> 10, c = i & 1023;
    int dir = c >> 9, cc = c & 511;
    out[off + i] = h32[dir * 16384 + b * 512 + cc];
}

// ---------------- launch ----------------
extern "C" void kernel_launch(void* const* d_in, const int* in_sizes, int n_in,
                              void* d_out, int out_size, void* d_ws, size_t ws_size,
                              hipStream_t stream) {
    (void)in_sizes; (void)n_in; (void)out_size; (void)ws_size;
    const float* inputs = (const float*)d_in[0];
    const float *K_[4][2], *R_[4][2], *Bw[4][2];
    for (int l = 0; l < 4; ++l)
        for (int dch = 0; dch < 2; ++dch) {
            int idx = 3 + l * 6 + dch * 3;
            K_[l][dch] = (const float*)d_in[idx];
            R_[l][dch] = (const float*)d_in[idx + 1];
            Bw[l][dch] = (const float*)d_in[idx + 2];
        }

    char* ws = (char*)d_ws;
    size_t off = 0;
    auto take = [&](size_t bytes) -> char* {
        char* p = ws + off;
        off += (bytes + 255) & ~(size_t)255;
        return p;
    };
    int* counters      = (int*)take(4 * sizeof(int));
    int* mask1         = (int*)take((size_t)32 * 1024 * 4);
    int* mask2         = (int*)take((size_t)32 * 512 * 4);
    int* mask3         = (int*)take((size_t)32 * 256 * 4);
    int* mask4         = (int*)take((size_t)32 * 128 * 4);
    _Float16* x1       = (_Float16*)take((size_t)32768 * 256 * 2);
    _Float16* kimg[4][2];
    for (int l = 0; l < 4; ++l)
        for (int dch = 0; dch < 2; ++dch) {
            size_t halves = (size_t)96 * ((l == 0) ? 8 : 64) * 512;
            kimg[l][dch] = (_Float16*)take(halves * 2);
        }
    _Float16* rimg     = (_Float16*)take((size_t)8 * 96 * 16 * 512 * 2);  // [l][dir]
    _Float16* xzbuf    = (_Float16*)take((size_t)2 * 32768 * 1536 * 2);
    _Float16* ys16     = (_Float16*)take((size_t)32 * 1024 * 1024 * 2);
    _Float16* h16      = (_Float16*)take((size_t)2 * 2 * 32 * 512 * 2);
    float*    h32      = (float*)take((size_t)2 * 32 * 512 * 4);

    (void)hipMemsetAsync(counters, 0, 16, stream);

    mask_from_input<<<(32 * 1024 + 255) / 256, 256, 0, stream>>>(inputs, mask1, 32 * 1024, 240);
    mask_down<<<(32 * 512 + 255) / 256, 256, 0, stream>>>(mask1, mask2, 512);
    mask_down<<<(32 * 256 + 255) / 256, 256, 0, stream>>>(mask2, mask3, 256);
    mask_down<<<(32 * 128 + 255) / 256, 256, 0, stream>>>(mask3, mask4, 128);

    pad_x_f16<<<(int)(((size_t)32768 * 256 + 255) / 256), 256, 0, stream>>>(inputs, x1, 32768, 240, 256);

    for (int l = 0; l < 4; ++l)
        for (int dch = 0; dch < 2; ++dch) {
            int kch = (l == 0) ? 8 : 64;
            int K = (l == 0) ? 240 : 2048;
            size_t tot = (size_t)96 * kch * 512;
            build_bimg<<<(int)((tot + 255) / 256), 256, 0, stream>>>(K_[l][dch], kimg[l][dch], kch, K);
            size_t rtot = (size_t)96 * 16 * 512;
            build_bimg<<<(int)((rtot + 255) / 256), 256, 0, stream>>>(
                R_[l][dch], rimg + ((size_t)(l * 2 + dch)) * rtot, 16, 512);
        }

    const int* masks[4] = {mask1, mask2, mask3, mask4};
    for (int l = 0; l < 4; ++l) {
        int T = 1024 >> l;
        int M = 32 * T;
        int kch = (l == 0) ? 8 : 64;
        int lda = (l == 0) ? 256 : 2048;
        const _Float16* Aptr = (l == 0) ? x1 : ys16;
        int tiles2 = (M / 16) * 48;   // each wave handles 2 ntiles
        for (int dch = 0; dch < 2; ++dch) {
            wmma_gemm_xz<<<tiles2 / 8, 256, 0, stream>>>(
                Aptr, lda, kch, kimg[l][dch], Bw[l][dch],
                xzbuf + (size_t)dch * M * 1536, M);
        }
        bigru_scan<<<64, 64, 0, stream>>>(
            rimg + (size_t)l * 2 * 96 * 16 * 512,
            Bw[l][0] + 1536, Bw[l][1] + 1536,
            xzbuf, masks[l], ys16,
            (l == 3) ? (float*)d_out : nullptr,
            h16, h32, T, counters + l);
    }

    assemble_hidden<<<(32 * 1024 + 255) / 256, 256, 0, stream>>>(
        h32, (float*)d_out, (size_t)32 * 128 * 1024);
}
